// MultiHeadAttn_32658931319413
// MI455X (gfx1250) — compile-verified, matrix operand order checked by reading, not run on hardware
//
#include <hip/hip_runtime.h>
#include <hip/hip_bf16.h>

// ---------------------------------------------------------------------------
// MI455X (gfx1250) MHA forward. All matrix math via v_wmma_f32_16x16x32_bf16.
//
// Fragment layout (CDNA5 ISA 7.12.2, 16-bit A 16x32; B stored as [N][K]):
//   lane l (l<16):  row = l,      K = kBase+[0..7]  and kBase+[16..23]
//   lane l (l>=16): row = l-16,   K = kBase+[8..15] and kBase+[24..31]
// C/D f32 16x16: vgpr r, lanes 0-15 -> (M=r, N=lane), lanes 16-31 -> (M=r+8).
// ---------------------------------------------------------------------------

typedef __bf16 bf16_t;
typedef __attribute__((ext_vector_type(16))) __bf16 v16bf;
typedef __attribute__((ext_vector_type(8)))  __bf16 bf16x8;
typedef __attribute__((ext_vector_type(4)))  __bf16 bf16x4;
typedef __attribute__((ext_vector_type(8)))  float  v8f;
typedef __attribute__((ext_vector_type(4)))  float  f32x4;
typedef __attribute__((ext_vector_type(4)))  unsigned int u32x4;

union FragU { v16bf v; bf16x8 h[2]; };

__device__ __forceinline__ v16bf load_frag(const bf16_t* base, int rowBase,
                                           int kBase, int stride) {
  const int lane = threadIdx.x & 31;
  const int l16  = lane & 15;
  const int half = lane >> 4;
  const bf16_t* p = base + (rowBase + l16) * stride + kBase + half * 8;
  FragU f;
  f.h[0] = *(const bf16x8*)(p);        // 16B -> K chunk 0
  f.h[1] = *(const bf16x8*)(p + 16);   // 16B -> K chunk 1
  return f.v;
}

__device__ __forceinline__ v8f wmma_bf16(v16bf a, v16bf b, v8f c) {
  return __builtin_amdgcn_wmma_f32_16x16x32_bf16(
      false, a, false, b, (short)0, c, false, false);
}

// ---------------------------------------------------------------------------
// fp32 -> bf16 conversion (vectorized x4) / 1024-row transpose (shift-based)
// ---------------------------------------------------------------------------
__global__ __launch_bounds__(256) void k_to_bf16_x4(const float* __restrict__ in,
                                                    bf16_t* __restrict__ out,
                                                    int n4) {
  int i = blockIdx.x * 256 + threadIdx.x;
  if (i >= n4) return;
  f32x4 v = *(const f32x4*)(in + i * 4);
  bf16x4 o;
  o.x = (bf16_t)v.x; o.y = (bf16_t)v.y; o.z = (bf16_t)v.z; o.w = (bf16_t)v.w;
  *(bf16x4*)(out + i * 4) = o;
}

// out[c][r] = (bf16) in[r][c]; in is 1024 x cols_in, out is cols_in x 1024.
__global__ __launch_bounds__(256) void k_transpose1024_bf16(
    const float* __restrict__ in, bf16_t* __restrict__ out, int cols_in, int n) {
  int i = blockIdx.x * 256 + threadIdx.x;
  if (i >= n) return;
  int c = i >> 10;          // / 1024 (rows_in == 1024 always)
  int r = i & 1023;
  out[i] = (bf16_t)in[r * cols_in + c];
}

// ---------------------------------------------------------------------------
// Shared GEMM mainloop: block tile 128(M) x 256(N) x 32(K), 256 threads =
// 8 waves laid out 2(M) x 4(N); wave tile 64x64 = 16 accumulators.
// 16 WMMA per 8 fragment loads per wave per k-step.
// A: [M][K] bf16 (lda=1024), Bt: [N][K] bf16 (ldb=1024), K total = 1024.
// ---------------------------------------------------------------------------
#define BK 32

__device__ __forceinline__ void gemm_mainloop(const bf16_t* __restrict__ A,
                                              const bf16_t* __restrict__ Bt,
                                              bf16_t* As, bf16_t* Bs,
                                              int bm, int bn, v8f acc[4][4]) {
  const int t = threadIdx.x;
  const int wave = t >> 5;
  const int wm = wave >> 2;   // 0..1
  const int wn = wave & 3;    // 0..3
  for (int kb = 0; kb < 1024; kb += BK) {
    // As: 128x32 -> 512 16B segments (2/thread). Bs: 256x32 -> 1024 (4/thread).
    for (int i = 0; i < 2; ++i) {
      int idx = t + i * 256;
      int row = idx >> 2, seg = idx & 3;
      *(u32x4*)(As + row * BK + seg * 8) =
          *(const u32x4*)(A + (bm * 128 + row) * 1024 + kb + seg * 8);
    }
    for (int i = 0; i < 4; ++i) {
      int idx = t + i * 256;
      int row = idx >> 2, seg = idx & 3;
      *(u32x4*)(Bs + row * BK + seg * 8) =
          *(const u32x4*)(Bt + (bn * 256 + row) * 1024 + kb + seg * 8);
    }
    __syncthreads();
    v16bf af[4];
    for (int mi = 0; mi < 4; ++mi) af[mi] = load_frag(As, wm * 64 + mi * 16, 0, BK);
    for (int ni = 0; ni < 4; ++ni) {
      v16bf bfr = load_frag(Bs, wn * 64 + ni * 16, 0, BK);
      for (int mi = 0; mi < 4; ++mi)
        acc[mi][ni] = wmma_bf16(af[mi], bfr, acc[mi][ni]);
    }
    __syncthreads();
  }
}

// ---------------------------------------------------------------------------
// GEMM 1: qkv = x @ w_qkv (M=8192, N=3072, K=1024); epilogue splits heads.
// grid (64, 12). (s,h) routing is wave-uniform per 16-wide column tile, and a
// 16-row M-tile never crosses a batch boundary, so each C tile is written from
// ONE base pointer with immediate offsets (r*stride).
// ---------------------------------------------------------------------------
__global__ __launch_bounds__(256)
__attribute__((amdgpu_waves_per_eu(1)))
void k_gemm_qkv(
    const bf16_t* __restrict__ A, const bf16_t* __restrict__ Bt,
    bf16_t* __restrict__ Qo, bf16_t* __restrict__ Ko, bf16_t* __restrict__ Vt) {
  __shared__ alignas(16) bf16_t As[128 * BK];
  __shared__ alignas(16) bf16_t Bs[256 * BK];
  const int bm = blockIdx.x, bn = blockIdx.y;
  v8f acc[4][4] = {};
  gemm_mainloop(A, Bt, As, Bs, bm, bn, acc);

  const int t = threadIdx.x;
  const int wave = t >> 5, wm = wave >> 2, wn = wave & 3;
  const int lane = t & 31, l16 = lane & 15, half = lane >> 4;
  for (int ni = 0; ni < 4; ++ni) {
    const int colBase = bn * 256 + wn * 64 + ni * 16;   // scalar per tile
    const int s   = colBase >> 10;                      // 0:Q 1:K 2:V
    const int rem = colBase & 1023;
    const int h   = rem >> 6;
    const int dd  = (rem & 63) + l16;                   // 0..63
    for (int mi = 0; mi < 4; ++mi) {
      const int rowBase = bm * 128 + wm * 64 + mi * 16; // 16-aligned
      const int b  = rowBase >> 11;                     // scalar per tile
      const int p0 = (rowBase & 2047) + 8 * half;
      const int bh = b * 16 + h;
      if (s == 0) {
        bf16_t* dst = Qo + (bh * 2048 + p0) * 64 + dd;
        #pragma unroll
        for (int r = 0; r < 8; ++r) dst[r * 64] = (bf16_t)acc[mi][ni][r];
      } else if (s == 1) {
        bf16_t* dst = Ko + (bh * 2048 + p0) * 64 + dd;
        #pragma unroll
        for (int r = 0; r < 8; ++r) dst[r * 64] = (bf16_t)acc[mi][ni][r];
      } else {
        bf16_t* dst = Vt + (bh * 64 + dd) * 2048 + p0;  // V transposed
        #pragma unroll
        for (int r = 0; r < 8; ++r) dst[r] = (bf16_t)acc[mi][ni][r];
      }
    }
  }
}

// ---------------------------------------------------------------------------
// Flash attention: 128 threads = 4 waves, one (b,h) per block, each wave owns
// a 16-row query tile. K/V blocks (32 keys) are staged cooperatively in LDS
// once per iteration and shared by all 4 waves (4x less global K/V traffic).
// Q,K: [b,h,p,64] bf16. Vt: [b,h,64,q] bf16. O: [b*2048+p][h*64+dd] bf16.
// ---------------------------------------------------------------------------
__global__ __launch_bounds__(128)
__attribute__((amdgpu_waves_per_eu(1)))
void k_attn(
    const bf16_t* __restrict__ Q, const bf16_t* __restrict__ K,
    const bf16_t* __restrict__ Vt, bf16_t* __restrict__ O) {
  __shared__ alignas(16) bf16_t Ks[32 * 64];       // 32 keys x 64 dims
  __shared__ alignas(16) bf16_t Vs[64 * 32];       // 64 dims x 32 keys (V^T)
  __shared__ alignas(16) bf16_t plds[4][16 * 32];  // per-wave P restage tile
  const int t = threadIdx.x;
  const int wave = t >> 5;
  const int idx = blockIdx.x;                // 0..2047 (b, h, 32 row-groups)
  const int b  = idx >> 9;
  const int rm = idx & 511;
  const int h  = rm >> 5;
  const int q0 = ((rm & 31) * 4 + wave) * 16;
  const int bh = b * 16 + h;
  const bf16_t* Qh = Q  + bh * 2048 * 64;
  const bf16_t* Kh = K  + bh * 2048 * 64;
  const bf16_t* Vh = Vt + bh * 64 * 2048;
  const int lane = t & 31, l16 = lane & 15, half = lane >> 4;
  bf16_t* pl = plds[wave];

  const v16bf qa0 = load_frag(Qh, q0, 0, 64);
  const v16bf qa1 = load_frag(Qh, q0, 32, 64);

  float m[8], lsum[8];
  v8f o[4] = {};
  for (int r = 0; r < 8; ++r) { m[r] = -INFINITY; lsum[r] = 0.0f; }

  for (int kb = 0; kb < 2048; kb += 32) {
    // Cooperative stage: Ks = K[kb..kb+31][0..63], Vs = Vt[0..63][kb..kb+31]
    for (int i = 0; i < 2; ++i) {
      int sidx = t + i * 128;                 // 256 x 16B segments
      int row = sidx >> 3, seg = sidx & 7;    // Ks: 8 segs/row
      *(u32x4*)(Ks + row * 64 + seg * 8) =
          *(const u32x4*)(Kh + (kb + row) * 64 + seg * 8);
    }
    for (int i = 0; i < 2; ++i) {
      int sidx = t + i * 128;                 // 256 x 16B segments
      int row = sidx >> 2, seg = sidx & 3;    // Vs: 4 segs/row
      *(u32x4*)(Vs + row * 32 + seg * 8) =
          *(const u32x4*)(Vh + row * 2048 + kb + seg * 8);
    }
    __syncthreads();

    v16bf k00 = load_frag(Ks, 0,  0,  64);
    v16bf k01 = load_frag(Ks, 0,  32, 64);
    v16bf k10 = load_frag(Ks, 16, 0,  64);
    v16bf k11 = load_frag(Ks, 16, 32, 64);
    v8f s0 = {}; s0 = wmma_bf16(qa0, k00, s0); s0 = wmma_bf16(qa1, k01, s0);
    v8f s1 = {}; s1 = wmma_bf16(qa0, k10, s1); s1 = wmma_bf16(qa1, k11, s1);

    float corr[8];
    for (int r = 0; r < 8; ++r) {
      float a = s0[r] * 0.125f;       // 1/sqrt(64)
      float c = s1[r] * 0.125f;
      float t0 = fmaxf(a, c);         // row max over the 16-lane half
      t0 = fmaxf(t0, __shfl_xor(t0, 1));
      t0 = fmaxf(t0, __shfl_xor(t0, 2));
      t0 = fmaxf(t0, __shfl_xor(t0, 4));
      t0 = fmaxf(t0, __shfl_xor(t0, 8));
      float mn = fmaxf(m[r], t0);
      corr[r] = __expf(m[r] - mn);
      m[r] = mn;
      float p0 = __expf(a - mn);
      float p1 = __expf(c - mn);
      float rs = p0 + p1;
      rs += __shfl_xor(rs, 1);
      rs += __shfl_xor(rs, 2);
      rs += __shfl_xor(rs, 4);
      rs += __shfl_xor(rs, 8);
      lsum[r] = lsum[r] * corr[r] + rs;
      int row = r + 8 * half;          // C layout -> row-major 16x32 LDS tile
      pl[row * 32 + l16]      = (bf16_t)p0;
      pl[row * 32 + 16 + l16] = (bf16_t)p1;
    }
    for (int ti = 0; ti < 4; ++ti)
      for (int r = 0; r < 8; ++r) o[ti][r] *= corr[r];

    v16bf pa = load_frag(pl, 0, 0, 32);
    v16bf v0 = load_frag(Vs, 0,  0, 32);
    v16bf v1 = load_frag(Vs, 16, 0, 32);
    v16bf v2 = load_frag(Vs, 32, 0, 32);
    v16bf v3 = load_frag(Vs, 48, 0, 32);
    o[0] = wmma_bf16(pa, v0, o[0]);
    o[1] = wmma_bf16(pa, v1, o[1]);
    o[2] = wmma_bf16(pa, v2, o[2]);
    o[3] = wmma_bf16(pa, v3, o[3]);
    __syncthreads();   // all waves done reading Ks/Vs before next stage
  }

  // One base pointer; r steps by 1024 bf16 rows, ti by 16 cols (immediates).
  bf16_t* dst = O + (b * 2048 + q0 + 8 * half) * 1024 + h * 64 + l16;
  for (int r = 0; r < 8; ++r) {
    float inv = 1.0f / lsum[r];
    #pragma unroll
    for (int ti = 0; ti < 4; ++ti)
      dst[r * 1024 + ti * 16] = (bf16_t)(o[ti][r] * inv);
  }
}

// ---------------------------------------------------------------------------
// GEMM 2: out = O @ w_proj + b (M=8192, N=1024, K=1024), fp32 output.
// grid (64, 4). One base pointer per C tile, r steps by 1024 floats.
// ---------------------------------------------------------------------------
__global__ __launch_bounds__(256)
__attribute__((amdgpu_waves_per_eu(1)))
void k_gemm_out(
    const bf16_t* __restrict__ A, const bf16_t* __restrict__ Bt,
    const float* __restrict__ bias, float* __restrict__ out) {
  __shared__ alignas(16) bf16_t As[128 * BK];
  __shared__ alignas(16) bf16_t Bs[256 * BK];
  const int bm = blockIdx.x, bn = blockIdx.y;
  v8f acc[4][4] = {};
  gemm_mainloop(A, Bt, As, Bs, bm, bn, acc);

  const int t = threadIdx.x;
  const int wave = t >> 5, wm = wave >> 2, wn = wave & 3;
  const int lane = t & 31, l16 = lane & 15, half = lane >> 4;
  for (int ni = 0; ni < 4; ++ni) {
    const int gcol = bn * 256 + wn * 64 + ni * 16 + l16;
    const float bv = bias[gcol];
    for (int mi = 0; mi < 4; ++mi) {
      const int rowBase = bm * 128 + wm * 64 + mi * 16 + 8 * half;
      float* dst = out + rowBase * 1024 + gcol;
      #pragma unroll
      for (int r = 0; r < 8; ++r) dst[r * 1024] = acc[mi][ni][r] + bv;
    }
  }
}

// ---------------------------------------------------------------------------
// Launcher. Workspace layout (bytes):
//   [0,16M)   xb   bf16 8192x1024
//   [16,22M)  wqT  bf16 3072x1024   (w_qkv transposed)
//   [22,24M)  wpT  bf16 1024x1024   (w_proj transposed)
//   [24,40M)  Q    bf16 [4,16,2048,64]
//   [40,56M)  K    bf16 [4,16,2048,64]
//   [56,72M)  Vt   bf16 [4,16,64,2048]
//   [72,88M)  Ob   bf16 8192x1024   (attention output, heads concatenated)
// ---------------------------------------------------------------------------
extern "C" void kernel_launch(void* const* d_in, const int* in_sizes, int n_in,
                              void* d_out, int out_size, void* d_ws, size_t ws_size,
                              hipStream_t stream) {
  const float* x      = (const float*)d_in[0];
  const float* w_qkv  = (const float*)d_in[1];
  const float* w_proj = (const float*)d_in[2];
  const float* b_proj = (const float*)d_in[3];
  float* out = (float*)d_out;

  char* ws = (char*)d_ws;
  bf16_t* xb  = (bf16_t*)(ws);
  bf16_t* wqT = (bf16_t*)(ws + (16u << 20));
  bf16_t* wpT = (bf16_t*)(ws + (22u << 20));
  bf16_t* Qb  = (bf16_t*)(ws + (24u << 20));
  bf16_t* Kb  = (bf16_t*)(ws + (40u << 20));
  bf16_t* Vt  = (bf16_t*)(ws + (56u << 20));
  bf16_t* Ob  = (bf16_t*)(ws + (72u << 20));

  const int nX4 = (4 * 2048 * 1024) / 4;   // 2097152 float4 groups
  const int nWq = 1024 * 3072;
  const int nWp = 1024 * 1024;

  k_to_bf16_x4<<<(nX4 + 255) / 256, 256, 0, stream>>>(x, xb, nX4);
  k_transpose1024_bf16<<<(nWq + 255) / 256, 256, 0, stream>>>(w_qkv, wqT, 3072, nWq);
  k_transpose1024_bf16<<<(nWp + 255) / 256, 256, 0, stream>>>(w_proj, wpT, 1024, nWp);

  k_gemm_qkv<<<dim3(64, 12), 256, 0, stream>>>(xb, wqT, Qb, Kb, Vt);

  k_attn<<<2048, 128, 0, stream>>>(Qb, Kb, Vt, Ob);

  k_gemm_out<<<dim3(64, 4), 256, 0, stream>>>(Ob, wpT, b_proj, out);
}
